// AttenSeq2Seq_65867618451668
// MI455X (gfx1250) — matrix operand order repaired and checked by reading, not compiled, not run on hardware
//
#include <hip/hip_runtime.h>
#include <hip/hip_bf16.h>
#include <math.h>

// CDNA5 (gfx1250) wave32 WMMA pipeline for AttenSeq2Seq sparse-conv attention.
typedef __attribute__((ext_vector_type(16))) __bf16 v16bf;
typedef __attribute__((ext_vector_type(8)))  __bf16 v8bf;
typedef __attribute__((ext_vector_type(8)))  float  v8f;

#define GRD 32
#define NVOX (GRD*GRD*GRD)
#define CH 96
#define NT 3              // output-channel tiles per wave (48 channels)

// ---------------- utility kernels ----------------
__global__ void k_zero(float* p, long long n) {
  long long i = (long long)blockIdx.x * blockDim.x + threadIdx.x;
  long long s = (long long)gridDim.x * blockDim.x;
  for (; i < n; i += s) p[i] = 0.f;
}

__global__ void k_points(const float* __restrict__ coords, int* __restrict__ flat,
                         float* __restrict__ cnt, int n) {
  int i = blockIdx.x * blockDim.x + threadIdx.x;
  if (i >= n) return;
  int vx = min(max((int)floorf(coords[i*3+0]), 0), GRD-1);
  int vy = min(max((int)floorf(coords[i*3+1]), 0), GRD-1);
  int vz = min(max((int)floorf(coords[i*3+2]), 0), GRD-1);
  int f = (vx*GRD + vy)*GRD + vz;
  flat[i] = f;
  atomicAdd(&cnt[f], 1.f);
}

__global__ void k_scatter(const float* __restrict__ feats, const float* __restrict__ scale,
                          const int* __restrict__ flat, float* __restrict__ gsum,
                          int Cin, int CinTot, int colOfs, long long total) {
  long long id = (long long)blockIdx.x * blockDim.x + threadIdx.x;
  if (id >= total) return;
  long long pt = id / Cin;
  int ch = (int)(id % Cin);
  float v = feats[id];
  if (scale) v *= scale[pt];
  atomicAdd(&gsum[(long long)flat[pt]*CinTot + colOfs + ch], v);
}

__global__ void k_normcvt(const float* __restrict__ gsum, const float* __restrict__ cnt,
                          __bf16* __restrict__ gbf, int Ctot, long long total) {
  long long id = (long long)blockIdx.x * blockDim.x + threadIdx.x;
  if (id >= total) return;
  long long vv = id / Ctot;
  gbf[id] = (__bf16)(gsum[id] / fmaxf(cnt[vv], 1.f));
}

// conv weights (O,I,3,3,3) f32 -> bf16 [tap][o][i]  (contiguous-K per output column)
__global__ void k_prep_conv(const float* __restrict__ w, __bf16* __restrict__ wbf,
                            int Cout, int Cin) {
  long long total = 27LL * Cout * Cin;
  long long id = (long long)blockIdx.x * blockDim.x + threadIdx.x;
  if (id >= total) return;
  int tap = (int)(id / ((long long)Cout * Cin));
  long long r = id % ((long long)Cout * Cin);
  int o = (int)(r / Cin), i = (int)(r % Cin);
  wbf[id] = (__bf16)(w[((long long)o*Cin + i)*27 + tap]);
}

__global__ void k_prep_lin(const float* __restrict__ w, __bf16* __restrict__ wbf, long long n) {
  long long id = (long long)blockIdx.x * blockDim.x + threadIdx.x;
  if (id >= n) return;
  wbf[id] = (__bf16)w[id];
}

// ---------------- implicit-GEMM 3x3x3 conv via WMMA ----------------
// One wave per 16-voxel x 48-outchannel tile: one A fragment feeds NT=3 WMMAs.
template<int CIN>
__global__ __launch_bounds__(32)
void k_conv_wmma(const __bf16* __restrict__ gbf, const __bf16* __restrict__ wbf,
                 float* __restrict__ gout) {
  const int base = blockIdx.x * 16;            // flat voxel base (z-contiguous)
  const int ncb  = blockIdx.y * (16*NT);       // output channel base
  const int x  = base >> 10;
  const int y  = (base >> 5) & (GRD-1);
  const int z0 = base & (GRD-1);
  const int lane = threadIdx.x;
  const int row  = lane & 15;                  // A: M row / B: N column held by this lane
  const int ksel = lane >> 4;                  // K half selector
  v8f acc[NT];
  #pragma unroll
  for (int t = 0; t < NT; ++t)
    #pragma unroll
    for (int v = 0; v < 8; ++v) acc[t][v] = 0.f;

  for (int tap = 0; tap < 27; ++tap) {
    const int dx = tap/9 - 1, dy = (tap/3)%3 - 1, dz = tap%3 - 1;
    if ((unsigned)(x+dx) >= GRD || (unsigned)(y+dy) >= GRD) continue;  // uniform (SGPR) branch
    const int nbase = base + dx*GRD*GRD + dy*GRD + dz;
    const bool rvalid = (unsigned)(z0 + row + dz) < (unsigned)GRD;     // per-lane z mask
    const __bf16* arow = gbf + (size_t)(nbase + row)*CIN + ksel*16;
    #pragma unroll
    for (int k0 = 0; k0 < CIN; k0 += 32) {
      v16bf a;
      if (rvalid) {
        v8bf lo = *(const v8bf*)(arow + k0);        // K = k0+ksel*16 .. +7
        v8bf hi = *(const v8bf*)(arow + k0 + 8);    // K = k0+ksel*16+8 .. +15
        #pragma unroll
        for (int j = 0; j < 8; ++j) { a[j] = lo[j]; a[8+j] = hi[j]; }
      } else {
        #pragma unroll
        for (int j = 0; j < 16; ++j) a[j] = (__bf16)0.f;
      }
      #pragma unroll
      for (int t = 0; t < NT; ++t) {
        const __bf16* wrow = wbf + ((size_t)tap*CH + (ncb + t*16 + row))*CIN;
        v16bf b = *(const v16bf*)(wrow + k0 + ksel*16);
        acc[t] = __builtin_amdgcn_wmma_f32_16x16x32_bf16(false, a, false, b, (short)0,
                                                         acc[t], false, false);
      }
    }
  }
  const int mo = (lane >> 4) * 8;
  #pragma unroll
  for (int t = 0; t < NT; ++t) {
    const int n = ncb + t*16 + (lane & 15);
    #pragma unroll
    for (int v = 0; v < 8; ++v)
      gout[(size_t)(base + mo + v)*CH + n] = acc[t][v];
  }
}

// ---------------- trilinear devoxelize (occupancy-masked) ----------------
__global__ void k_devox(const float* __restrict__ coords, const float* __restrict__ cnt,
                        const float* __restrict__ gout, float* __restrict__ dv,
                        int Cout, long long total) {
  long long id = (long long)blockIdx.x * blockDim.x + threadIdx.x;
  if (id >= total) return;
  int c = (int)(id % Cout);
  long long pt = id / Cout;
  float cx = coords[pt*3+0], cy = coords[pt*3+1], cz = coords[pt*3+2];
  float bx = floorf(cx), by = floorf(cy), bz = floorf(cz);
  float fx = cx-bx, fy = cy-by, fz = cz-bz;
  int bxi = (int)bx, byi = (int)by, bzi = (int)bz;
  float acc = 0.f;
  #pragma unroll
  for (int corner = 0; corner < 8; ++corner) {
    int dx = (corner>>2)&1, dy = (corner>>1)&1, dz = corner&1;
    int ix = bxi+dx, iy = byi+dy, iz = bzi+dz;
    bool valid = (unsigned)ix < GRD && (unsigned)iy < GRD && (unsigned)iz < GRD;
    int cxi = min(max(ix,0),GRD-1), cyi = min(max(iy,0),GRD-1), czi = min(max(iz,0),GRD-1);
    int fl = (cxi*GRD + cyi)*GRD + czi;
    float w = (dx ? fx : 1.f-fx) * (dy ? fy : 1.f-fy) * (dz ? fz : 1.f-fz);
    if (valid && cnt[fl] > 0.f) acc += gout[(size_t)fl*Cout + c] * w;
  }
  dv[id] = acc;
}

// -------- fused pointwise linear via WMMA: out = tanh?(devox + bias + A @ W^T) --------
// A channels: i<CH -> srcA, else srcB (optionally scaled per point) == concat, resolved
// at compile time per unrolled K step (8-wide groups never straddle the CH boundary).
template<int CIN>
__global__ __launch_bounds__(32)
void k_linear_wmma(const float* __restrict__ srcA, const float* __restrict__ srcB,
                   const float* __restrict__ scaleB,
                   const __bf16* __restrict__ lwbf, const float* __restrict__ bias,
                   const float* __restrict__ devox, float* __restrict__ out,
                   int doTanh, int n) {
  const int m0  = blockIdx.x * 16;
  const int ncb = blockIdx.y * (16*NT);
  const int lane = threadIdx.x;
  const int row  = lane & 15;
  const int ksel = lane >> 4;
  const int m    = m0 + row;                   // A row for this lane
  const int mc   = m < n ? m : n - 1;
  const int mo   = (lane >> 4) * 8;
  v8f acc[NT];
  #pragma unroll
  for (int t = 0; t < NT; ++t) {
    const int ncol = ncb + t*16 + (lane & 15);
    #pragma unroll
    for (int v = 0; v < 8; ++v) {
      int mm = min(m0 + mo + v, n - 1);
      acc[t][v] = devox[(size_t)mm*CH + ncol] + bias[ncol];
    }
  }
  #pragma unroll
  for (int k0 = 0; k0 < CIN; k0 += 32) {
    v16bf a;
    #pragma unroll
    for (int half = 0; half < 2; ++half) {
      const bool second = (CIN == 2*CH) && (k0 + half*16 >= CH);  // compile-time per step
      const float* src = second ? srcB : srcA;
      const int iofs = (k0 + half*8) - (second ? CH : 0) + ksel*16;
      float sc = (second && scaleB) ? scaleB[mc] : 1.f;
      #pragma unroll
      for (int j = 0; j < 8; ++j)
        a[half*8 + j] = (__bf16)(src[(size_t)mc*CH + iofs + j] * sc);
    }
    #pragma unroll
    for (int t = 0; t < NT; ++t) {
      const __bf16* wrow = lwbf + (size_t)(ncb + t*16 + row) * CIN;
      v16bf b = *(const v16bf*)(wrow + k0 + ksel*16);
      acc[t] = __builtin_amdgcn_wmma_f32_16x16x32_bf16(false, a, false, b, (short)0,
                                                       acc[t], false, false);
    }
  }
  #pragma unroll
  for (int t = 0; t < NT; ++t) {
    const int ncol = ncb + t*16 + (lane & 15);
    #pragma unroll
    for (int v = 0; v < 8; ++v) {
      int mm = m0 + mo + v;
      if (mm < n) {
        float r = acc[t][v];
        if (doTanh) r = tanhf(r);
        out[(size_t)mm*CH + ncol] = r;
      }
    }
  }
}

// ---------------- Cout=1 conv (attention scalar) + fused attn eval ----------------
__global__ void k_conv_v(const __bf16* __restrict__ gbf, const __bf16* __restrict__ wvbf,
                         float* __restrict__ gout1) {
  int v = blockIdx.x * blockDim.x + threadIdx.x;
  if (v >= NVOX) return;
  int x = v >> 10, y = (v >> 5) & (GRD-1), z = v & (GRD-1);
  float acc = 0.f;
  for (int tap = 0; tap < 27; ++tap) {
    int dx = tap/9 - 1, dy = (tap/3)%3 - 1, dz = tap%3 - 1;
    if ((unsigned)(x+dx) >= GRD || (unsigned)(y+dy) >= GRD || (unsigned)(z+dz) >= GRD) continue;
    int nb = v + dx*GRD*GRD + dy*GRD + dz;
    for (int i = 0; i < CH; ++i)
      acc += (float)gbf[(size_t)nb*CH + i] * (float)wvbf[tap*CH + i];
  }
  gout1[v] = acc;
}

__global__ void k_attn(const float* __restrict__ energy, const float* __restrict__ vlw,
                       const float* __restrict__ vlb, const float* __restrict__ coords,
                       const float* __restrict__ cnt, const float* __restrict__ gout1,
                       float* __restrict__ attn, int n) {
  int pt = blockIdx.x * blockDim.x + threadIdx.x;
  if (pt >= n) return;
  float lin = vlb[0];
  for (int i = 0; i < CH; ++i) lin += energy[(size_t)pt*CH + i] * vlw[i];
  float cx = coords[pt*3+0], cy = coords[pt*3+1], cz = coords[pt*3+2];
  float bx = floorf(cx), by = floorf(cy), bz = floorf(cz);
  float fx = cx-bx, fy = cy-by, fz = cz-bz;
  int bxi = (int)bx, byi = (int)by, bzi = (int)bz;
  float dvv = 0.f;
  #pragma unroll
  for (int corner = 0; corner < 8; ++corner) {
    int dx = (corner>>2)&1, dy = (corner>>1)&1, dz = corner&1;
    int ix = bxi+dx, iy = byi+dy, iz = bzi+dz;
    bool valid = (unsigned)ix < GRD && (unsigned)iy < GRD && (unsigned)iz < GRD;
    int cxi = min(max(ix,0),GRD-1), cyi = min(max(iy,0),GRD-1), czi = min(max(iz,0),GRD-1);
    int fl = (cxi*GRD + cyi)*GRD + czi;
    float w = (dx ? fx : 1.f-fx) * (dy ? fy : 1.f-fy) * (dz ? fz : 1.f-fz);
    if (valid && cnt[fl] > 0.f) dvv += gout1[fl] * w;
  }
  attn[pt] = lin + dvv;
}

// ---------------- host orchestration ----------------
extern "C" void kernel_launch(void* const* d_in, const int* in_sizes, int n_in,
                              void* d_out, int out_size, void* d_ws, size_t ws_size,
                              hipStream_t stream) {
  const int N = in_sizes[0] / CH;
  const float* hidden = (const float*)d_in[0];
  const float* query  = (const float*)d_in[1];
  const float* coords = (const float*)d_in[2];
  const float* wq_conv = (const float*)d_in[3];
  const float* wq_lw = (const float*)d_in[4];  const float* wq_lb = (const float*)d_in[5];
  const float* wh_conv = (const float*)d_in[6];
  const float* wh_lw = (const float*)d_in[7];  const float* wh_lb = (const float*)d_in[8];
  const float* att_conv = (const float*)d_in[9];
  const float* att_lw = (const float*)d_in[10]; const float* att_lb = (const float*)d_in[11];
  const float* v_conv = (const float*)d_in[12];
  const float* v_lw = (const float*)d_in[13];   const float* v_lb = (const float*)d_in[14];
  const float* upd_conv = (const float*)d_in[15];
  const float* upd_lw = (const float*)d_in[16]; const float* upd_lb = (const float*)d_in[17];

  // workspace layout (256B aligned slices)
  char* ws = (char*)d_ws;
  size_t off = 0;
  auto take = [&](size_t bytes) -> char* {
    char* p = ws + off;
    off = (off + bytes + 255) & ~(size_t)255;
    return p;
  };
  int*    flat  = (int*)   take((size_t)N * 4);
  float*  cnt   = (float*) take((size_t)NVOX * 4);
  float*  gsum  = (float*) take((size_t)NVOX * 2*CH * 4);
  __bf16* gbf   = (__bf16*)take((size_t)NVOX * 2*CH * 2);
  float*  gout  = (float*) take((size_t)NVOX * CH * 4);
  float*  dv    = (float*) take((size_t)N * CH * 4);
  float*  qbuf  = (float*) take((size_t)N * CH * 4);
  float*  hbuf  = (float*) take((size_t)N * CH * 4);
  float*  ebuf  = (float*) take((size_t)N * CH * 4);
  float*  attn  = (float*) take((size_t)N * 4);
  __bf16* wcbf  = (__bf16*)take((size_t)27 * CH * 2*CH * 2);
  __bf16* wlbf  = (__bf16*)take((size_t)CH * 2*CH * 2);

  const int T = 256;
  auto blk = [&](long long tot) { return (unsigned)((tot + T - 1) / T); };
  const dim3 convGrid(NVOX/16, CH/(16*NT)), gemmGrid((N + 15)/16, CH/(16*NT));

  // shared voxel indexing + occupancy counts
  k_zero<<<blk(NVOX), T, 0, stream>>>(cnt, NVOX);
  k_points<<<blk(N), T, 0, stream>>>(coords, flat, cnt, N);

  // generic sconv stage (Cout = CH) as a sequence
  auto sconv = [&](const float* fA, const float* fB, const float* scB, int Cin,
                   const float* wconv, const float* lw, const float* lb,
                   float* outbuf, int doTanh) {
    long long gels = (long long)NVOX * Cin;
    k_zero<<<blk(gels), T, 0, stream>>>(gsum, gels);
    k_scatter<<<blk((long long)N*CH), T, 0, stream>>>(fA, nullptr, flat, gsum, CH, Cin, 0,
                                                      (long long)N*CH);
    if (fB)
      k_scatter<<<blk((long long)N*CH), T, 0, stream>>>(fB, scB, flat, gsum, CH, Cin, CH,
                                                        (long long)N*CH);
    k_normcvt<<<blk(gels), T, 0, stream>>>(gsum, cnt, gbf, Cin, gels);
    k_prep_conv<<<blk(27LL*CH*Cin), T, 0, stream>>>(wconv, wcbf, CH, Cin);
    if (Cin == CH)
      k_conv_wmma<CH><<<convGrid, 32, 0, stream>>>(gbf, wcbf, gout);
    else
      k_conv_wmma<2*CH><<<convGrid, 32, 0, stream>>>(gbf, wcbf, gout);
    k_devox<<<blk((long long)N*CH), T, 0, stream>>>(coords, cnt, gout, dv, CH, (long long)N*CH);
    k_prep_lin<<<blk((long long)CH*Cin), T, 0, stream>>>(lw, wlbf, (long long)CH*Cin);
    if (Cin == CH)
      k_linear_wmma<CH><<<gemmGrid, 32, 0, stream>>>(fA, fB, scB, wlbf, lb, dv, outbuf,
                                                     doTanh, N);
    else
      k_linear_wmma<2*CH><<<gemmGrid, 32, 0, stream>>>(fA, fB, scB, wlbf, lb, dv, outbuf,
                                                       doTanh, N);
  };

  // q = sconv(query); h = sconv(hidden)
  sconv(query,  nullptr, nullptr, CH, wq_conv, wq_lw, wq_lb, qbuf, 0);
  sconv(hidden, nullptr, nullptr, CH, wh_conv, wh_lw, wh_lb, hbuf, 0);
  // energy = tanh(sconv(concat(h, q)))
  sconv(hbuf, qbuf, nullptr, 2*CH, att_conv, att_lw, att_lb, ebuf, 1);

  // attn = sconv_v(energy): Cout = 1 (VALU path, too skinny for WMMA)
  {
    long long gels = (long long)NVOX * CH;
    k_zero<<<blk(gels), T, 0, stream>>>(gsum, gels);
    k_scatter<<<blk((long long)N*CH), T, 0, stream>>>(ebuf, nullptr, flat, gsum, CH, CH, 0,
                                                      (long long)N*CH);
    k_normcvt<<<blk(gels), T, 0, stream>>>(gsum, cnt, gbf, CH, gels);
    k_prep_conv<<<blk(27LL*CH), T, 0, stream>>>(v_conv, wcbf, 1, CH);
    k_conv_v<<<blk(NVOX), T, 0, stream>>>(gbf, wcbf, gout);
    k_attn<<<blk(N), T, 0, stream>>>(ebuf, v_lw, v_lb, coords, cnt, gout, attn, N);
  }

  // out = tanh(sconv(concat(q, attn * h)))
  sconv(qbuf, hbuf, attn, 2*CH, upd_conv, upd_lw, upd_lb, (float*)d_out, 1);
}
